// SSIMLoss_27195732919032
// MI455X (gfx1250) — matrix operand order repaired and checked by reading, not compiled, not run on hardware
//
#include <hip/hip_runtime.h>
#include <hip/hip_bf16.h>

typedef __attribute__((ext_vector_type(16))) _Float16 v16h;
typedef __attribute__((ext_vector_type(8)))  _Float16 v8h;
typedef __attribute__((ext_vector_type(8)))  float    v8f;

#define C1_SSIM 0.0001f   // (0.01*1.0)^2
#define C2_SSIM 0.0009f   // (0.03*1.0)^2

// tile geometry: 64x64 outputs, 74x74 halo, padded to 80x80 in LDS
#define IN_PITCH   80
#define IN_MAP     (80*80)          // 6400 f16 per map
#define MID_PITCH  80               // transposed: [col][row], 80 rows
#define MID_MAP    (64*80)          // 5120 f16 per map
// 5 in-maps + 5 mid-maps (f16) + 16 floats for the 1-D Gaussian (+zero sentinel)
#define LDS_BYTES  ((5*IN_MAP + 5*MID_MAP) * 2 + 16 * 4)   // 115264 B

__device__ __forceinline__ v16h cat8(v8h lo, v8h hi) {
  return __builtin_shufflevector(lo, hi, 0,1,2,3,4,5,6,7,8,9,10,11,12,13,14,15);
}

__global__ __launch_bounds__(256) void ssim_tile_kernel(
    const float* __restrict__ pred, const float* __restrict__ target,
    const float* __restrict__ k2d, float* __restrict__ blockSums)
{
  extern __shared__ char smem_raw[];
  _Float16* sIn  = (_Float16*)smem_raw;      // 5 maps, [row][col] 80x80
  _Float16* sMid = sIn + 5*IN_MAP;           // 5 maps, transposed [col][row] 64x80
  float*    shg  = (float*)(sMid + 5*MID_MAP); // g[0..10], shg[11] = 0 sentinel
  float*    sRed = (float*)smem_raw;         // reused for reduction (sIn dead by then)

  const int tid  = threadIdx.x;
  const int lane = tid & 31;
  const int wave = tid >> 5;
  const int ln   = lane & 15;                // M (A-row) / N (B-col) per layout tables
  const int half = lane >> 4;

  // Recover separable 1-D Gaussian from k2d[5][j] = g[5]*g[j]; stash in LDS so the
  // band build below is ds_load + clamp instead of an 11-way v_cndmask cascade.
  if (tid < 12) {
    shg[tid] = (tid < 11) ? k2d[5*11 + tid] * rsqrtf(k2d[5*11 + 5]) : 0.0f;
  }

  // Tile placement: blockIdx = img*64 + tile, 8x8 tiles of 64x64 per 512x512 image
  const int img  = blockIdx.x >> 6;          // n*3 + c, 0..95
  const int tile = blockIdx.x & 63;
  const int row0 = ((tile >> 3) << 6) - 5;   // halo origin (zero pad = 5)
  const int col0 = ((tile & 7) << 6) - 5;
  const int base = img * (512*512);

  // Load 74x74 halo, zero-pad to 80x80, build the five maps in f16
  for (int idx = tid; idx < IN_MAP; idx += 256) {
    int lr = idx / IN_PITCH, lc = idx - lr*IN_PITCH;
    float x = 0.f, y = 0.f;
    if (lr < 74 && lc < 74) {
      int gr = row0 + lr, gc = col0 + lc;
      if (gr >= 0 && gr < 512 && gc >= 0 && gc < 512) {
        int off = base + gr*512 + gc;
        x = pred[off]; y = target[off];
      }
    }
    sIn[0*IN_MAP + idx] = (_Float16)x;
    sIn[1*IN_MAP + idx] = (_Float16)y;
    sIn[2*IN_MAP + idx] = (_Float16)(x*x);
    sIn[3*IN_MAP + idx] = (_Float16)(y*y);
    sIn[4*IN_MAP + idx] = (_Float16)(x*y);
  }
  __syncthreads();

  // Constant banded weight operands (B: ISA 7.12.2 table k order; A: contiguous k, same as B_mid):
  //  B 32x16 f16: element j -> K = 8*half + j (+8 if j>=8), lane&15 = N. B[k][n]=g[k-n]
  //  A 16x32 f16: element j -> K = 16*half + j,            lane&15 = M. A[m][k]=g[k-m]
  // Branchless: clamp out-of-band offsets to the zero sentinel shg[11].
  v16h bandB, bandA;
  #pragma unroll
  for (int j = 0; j < 16; ++j) {
    unsigned kb = (unsigned)(8*half + j + ((j >= 8) ? 8 : 0) - ln);
    bandB[j] = (_Float16)shg[kb <= 10u ? kb : 11u];
    unsigned ka = (unsigned)(16*half + j - ln);
    bandA[j] = (_Float16)shg[ka <= 10u ? ka : 11u];
  }

  // ---- Stage 1: horizontal 11-tap conv as D(16x16) = A_data(16x32) x B_band(32x16)
  // 5 row-groups (rows 0..79, pad rows are zero) x 4 col-groups x 5 maps = 100 WMMAs
  for (int t = wave; t < 100; t += 8) {
    int map = t % 5, pos = t / 5;
    int rg = pos >> 2, cg = pos & 3;
    const _Float16* p = sIn + map*IN_MAP + (rg*16 + ln)*IN_PITCH + cg*16 + 8*half;
    v16h a = cat8(*(const v8h*)p, *(const v8h*)(p + 16));   // K 0-7/16-23 | 8-15/24-31
    v8f c = {};
    c = __builtin_amdgcn_wmma_f32_16x16x32_f16(false, a, false, bandB,
                                               (short)0, c, false, false);
    // D element i -> out row rg*16 + i + 8*half, col cg*16 + ln; store transposed
    v8h o;
    #pragma unroll
    for (int i = 0; i < 8; ++i) o[i] = (_Float16)c[i];
    *(v8h*)(sMid + map*MID_MAP + (cg*16 + ln)*MID_PITCH + rg*16 + 8*half) = o;
  }
  __syncthreads();

  // ---- Stage 2: vertical conv as D = A_band(16x32) x B_mid(32x16), then SSIM
  float partial = 0.f;
  for (int s = wave; s < 16; s += 8) {                      // 16 subtiles of 16x16
    int sr = s >> 2, sc = s & 3;
    v8f acc[5];
    #pragma unroll
    for (int map = 0; map < 5; ++map) {
      // B element j -> row K = sr*16 + 16*half + j (contiguous in transposed mid)
      const _Float16* p = sMid + map*MID_MAP + (sc*16 + ln)*MID_PITCH + sr*16 + 16*half;
      v16h b = cat8(*(const v8h*)p, *(const v8h*)(p + 8));
      v8f c = {};
      acc[map] = __builtin_amdgcn_wmma_f32_16x16x32_f16(false, bandA, false, b,
                                                        (short)0, c, false, false);
    }
    #pragma unroll
    for (int i = 0; i < 8; ++i) {
      float mx  = acc[0][i], my  = acc[1][i];
      float ex2 = acc[2][i], ey2 = acc[3][i], exy = acc[4][i];
      float mx2 = mx*mx, my2 = my*my, mxy = mx*my;
      float num = (2.f*mxy + C1_SSIM) * (2.f*(exy - mxy) + C2_SSIM);
      float den = (mx2 + my2 + C1_SSIM) * ((ex2 - mx2) + (ey2 - my2) + C2_SSIM);
      partial += num / den;
    }
  }

  // Block reduction (sIn region is dead; safe to alias)
  __syncthreads();
  sRed[tid] = partial;
  __syncthreads();
  #pragma unroll
  for (int k = 128; k > 0; k >>= 1) {
    if (tid < k) sRed[tid] += sRed[tid + k];
    __syncthreads();
  }
  if (tid == 0) blockSums[blockIdx.x] = sRed[0];
}

// Deterministic final reduction: fixed partition + fixed tree order, no atomics.
__global__ __launch_bounds__(256) void ssim_finalize_kernel(
    const float* __restrict__ bs, float* __restrict__ out, int nb)
{
  __shared__ float red[256];
  int tid = threadIdx.x;
  float s = 0.f;
  for (int i = tid; i < nb; i += 256) s += bs[i];
  red[tid] = s;
  __syncthreads();
  #pragma unroll
  for (int k = 128; k > 0; k >>= 1) {
    if (tid < k) red[tid] += red[tid + k];
    __syncthreads();
  }
  if (tid == 0) out[0] = 1.0f - red[0] * (1.0f / 25165824.0f);  // N = 32*3*512*512
}

extern "C" void kernel_launch(void* const* d_in, const int* in_sizes, int n_in,
                              void* d_out, int out_size, void* d_ws, size_t ws_size,
                              hipStream_t stream) {
  (void)in_sizes; (void)n_in; (void)out_size; (void)ws_size;
  const float* pred   = (const float*)d_in[0];
  const float* target = (const float*)d_in[1];
  const float* k2d    = (const float*)d_in[2];
  float* out       = (float*)d_out;
  float* blockSums = (float*)d_ws;          // 6144 floats = 24 KB of scratch

  const int NB = 96 * 64;                   // (32*3 images) x (8x8 tiles)
  ssim_tile_kernel<<<NB, 256, LDS_BYTES, stream>>>(pred, target, k2d, blockSums);
  ssim_finalize_kernel<<<1, 256, 0, stream>>>(blockSums, out, NB);
}
